// MemoryReader_2370821947394
// MI455X (gfx1250) — matrix operand order repaired and checked by reading, not compiled, not run on hardware
//
#include <hip/hip_runtime.h>
#include <hip/hip_bf16.h>
#include <math.h>

// ---------------------------------------------------------------------------
// MemoryReader readout as flash-attention on CDNA5 WMMA (gfx1250, wave32).
//   B=4, CK=64, CV=512, THW=12960, HW=1620
//   score[t,q] = 0.25*(mk.qk)[t,q] - 0.125*|mk[:,t]|^2   (sqrt(CK)=8)
//   P = softmax_t(score);  out[:,0:512] = mv @ P;  out[:,512:1024] = qv
//
// Round 3: keep the packed bf16 WMMA-fragment pipeline; stream the dominant
// mv fragment traffic through the Tensor Data Mover into per-wave LDS double
// buffers (tensor_load_to_lds + s_wait_tensorcnt), overlapping TDM DMA of
// chunk i+1 with WMMA on chunk i. Falls back to direct global b128 loads if
// the TDM builtin is unavailable.
// ---------------------------------------------------------------------------

#define BB   4
#define CK   64
#define CV   512
#define THW  12960
#define HW   1620
#define QT   16            // queries per block tile
#define NQT  102           // ceil(HW/16)
#define NST  810           // THW/16  (exact)
#define NKT  203           // ceil(THW/64)
#define NTK  405           // THW/32  (exact): K=32 fragments along t
#define NTK_PAD 406        // +1 zero pad fragment so TDM tiles never go OOB
#define FRAG_ELEMS 512     // 32 lanes * 16 bf16 per fragment (1024 B)

typedef __attribute__((ext_vector_type(16))) __bf16 v16bf;
typedef __attribute__((ext_vector_type(8)))  __bf16 v8bf;
typedef __attribute__((ext_vector_type(8)))  float  v8f;
typedef __attribute__((ext_vector_type(4)))  unsigned int u32x4;
typedef __attribute__((ext_vector_type(8)))  int i32x8;
typedef __attribute__((ext_vector_type(4)))  int i32x4;

#define WMMA_BF16(A, Bm, C) \
  __builtin_amdgcn_wmma_f32_16x16x32_bf16(false, (A), false, (Bm), (short)0, (C), false, false)

#if __has_builtin(__builtin_amdgcn_tensor_load_to_lds) && \
    __has_builtin(__builtin_amdgcn_s_wait_tensorcnt)
#define USE_TDM 1
#else
#define USE_TDM 0
#endif

// A (16x32 bf16) packing used here (contiguous K, same in B): M = lane&15, h = lane>>4, VGPR j holds K pair:
//   K = 16h + 2j   (vector elems 2j,2j+1)
__device__ __forceinline__ int a_kpair(int j, int half) {
  return 16 * half + 2 * j;
}

#if USE_TDM
// ---------------------------------------------------------------------------
// Issue one TDM 2D tile load: tile_d1 rows x tile_d0 8-byte elements,
// row stride row_stride_elems, global -> LDS. D# packing per ISA 8.3/8.4.
// All inputs are wave-uniform -> descriptor lives in SGPRs.
// ---------------------------------------------------------------------------
__device__ __forceinline__ void tdm_load_2d(unsigned lds_addr,
                                            unsigned long long gaddr,
                                            unsigned tile_d0, unsigned tile_d1,
                                            unsigned long long row_stride_elems) {
  u32x4 g0;
  g0[0] = 1u;                                            // count=1, user D#
  g0[1] = lds_addr;                                      // LDS byte address
  g0[2] = (unsigned)(gaddr & 0xffffffffu);               // global_addr[31:0]
  g0[3] = (unsigned)((gaddr >> 32) & 0x01ffffffu)        // global_addr[56:32]
        | 0x80000000u;                                   // type=2 ("image")
  i32x8 g1 = {};
  g1[0] = (3 << 16);                                     // data_size=8B, mask=0
  g1[1] = (int)((tile_d0 & 0xffffu) << 16);              // tensor_dim0 lo16
  g1[2] = (int)((tile_d0 >> 16) & 0xffffu)               // tensor_dim0 hi16
        | (int)((tile_d1 & 0xffffu) << 16);              // tensor_dim1 lo16
  g1[3] = (int)((tile_d1 >> 16) & 0xffffu)               // tensor_dim1 hi16
        | (int)((tile_d0 & 0xffffu) << 16);              // tile_dim0
  g1[4] = (int)(tile_d1 & 0xffffu);                      // tile_dim1 (dim2=0)
  g1[5] = (int)(row_stride_elems & 0xffffffffu);         // dim0_stride lo32
  g1[6] = (int)((row_stride_elems >> 32) & 0xffffu);     // dim0_stride hi16
  g1[7] = 0;
  i32x4 z4 = {};
#if defined(__clang_major__) && (__clang_major__ >= 23)
  i32x8 z8 = {};
  __builtin_amdgcn_tensor_load_to_lds(g0, g1, z4, z4, z8, 0);
#else
  __builtin_amdgcn_tensor_load_to_lds(g0, g1, z4, z4, 0);
#endif
}
#endif // USE_TDM

// ---------------------------------------------------------------------------
// Kernel 0: msq[b,t] = sum_c mk[b,c,t]^2
// ---------------------------------------------------------------------------
__global__ __launch_bounds__(256) void msq_kernel(const float* __restrict__ mk,
                                                  float* __restrict__ msq) {
  int i = blockIdx.x * 256 + threadIdx.x;
  if (i >= BB * THW) return;
  int b = i / THW, t = i - b * THW;
  const float* p = mk + (size_t)b * CK * THW + t;
  float s = 0.f;
#pragma unroll 8
  for (int c = 0; c < CK; ++c) { float v = p[(size_t)c * THW]; s += v * v; }
  msq[i] = s;
}

// ---------------------------------------------------------------------------
// Kernel 1: pack mk into bf16 A-fragments.
// Fragment id = (b*NST + si)*2 + f  (si: 16-row t-subtile, f: K=32 c-half).
// Grid exact: BB*NST*2*32 threads = 810 blocks x 256.
// ---------------------------------------------------------------------------
__global__ __launch_bounds__(256) void pack_mk_kernel(const float* __restrict__ mk,
                                                      __bf16* __restrict__ mkp) {
  int tid = blockIdx.x * 256 + threadIdx.x;
  int fid = tid >> 5, lane = tid & 31;
  int b = fid / (NST * 2);
  int rem = fid - b * NST * 2;
  int si = rem >> 1, f = rem & 1;
  int half = lane >> 4;
  int t = si * 16 + (lane & 15);
  const float* mkb = mk + (size_t)b * CK * THW;
  v16bf a;
#pragma unroll
  for (int j = 0; j < 8; ++j) {
    int kp = a_kpair(j, half);
    int c = 32 * f + kp;
    a[2 * j]     = (__bf16)mkb[(size_t)c * THW + t];
    a[2 * j + 1] = (__bf16)mkb[(size_t)(c + 1) * THW + t];
  }
  *(v16bf*)(mkp + (size_t)fid * FRAG_ELEMS + lane * 16) = a;
}

// ---------------------------------------------------------------------------
// Kernel 2: pack mv into bf16 A-fragments (padded rows of NTK_PAD fragments).
// Fragment id = (b*32 + ctile)*NTK_PAD + tk; tk==NTK writes a zero fragment.
// Grid exact: BB*32*NTK_PAD*32 threads = 6496 blocks x 256.
// ---------------------------------------------------------------------------
__global__ __launch_bounds__(256) void pack_mv_kernel(const float* __restrict__ mv,
                                                      __bf16* __restrict__ mvp) {
  int tid = blockIdx.x * 256 + threadIdx.x;
  int fid = tid >> 5, lane = tid & 31;
  int b = fid / (32 * NTK_PAD);
  int rem = fid - b * 32 * NTK_PAD;
  int ct = rem / NTK_PAD, tk = rem - ct * NTK_PAD;
  v16bf a = {};
  if (tk < NTK) {
    int half = lane >> 4;
    int c = ct * 16 + (lane & 15);
    const float* mvr = mv + ((size_t)b * CV + c) * THW + tk * 32;
#pragma unroll
    for (int j = 0; j < 8; ++j) {
      int kp = a_kpair(j, half);               // always even
      float2 p = *(const float2*)(mvr + kp);
      a[2 * j]     = (__bf16)p.x;
      a[2 * j + 1] = (__bf16)p.y;
    }
  }
  *(v16bf*)(mvp + (size_t)fid * FRAG_ELEMS + lane * 16) = a;
}

// ---------------------------------------------------------------------------
// Kernel 3: per-(b, q-tile) softmax statistics over the memory axis.
// ---------------------------------------------------------------------------
__global__ __launch_bounds__(256) void pass_stats(const __bf16* __restrict__ mkp,
                                                  const float* __restrict__ qk,
                                                  const float* __restrict__ msq,
                                                  float* __restrict__ stat_m,
                                                  float* __restrict__ stat_s) {
  const int qt = blockIdx.x, b = blockIdx.y;
  const int lane = threadIdx.x & 31, w = threadIdx.x >> 5;
  const int qcol = lane & 15, half = lane >> 4;
  const int qg = min(qt * QT + qcol, HW - 1);

  const float* qkb  = qk  + (size_t)b * CK * HW;
  const float* msqb = msq + (size_t)b * THW;

  v16bf bq0, bq1;
#pragma unroll
  for (int e = 0; e < 16; ++e) {
    int c = 16 * half + e;
    bq0[e] = (__bf16)qkb[(size_t)c * HW + qg];
    bq1[e] = (__bf16)qkb[(size_t)(c + 32) * HW + qg];
  }

  float m_run = -3.0e38f, s_run = 0.0f;

  for (int si = w; si < NST; si += 8) {
    const size_t base = ((size_t)(b * NST + si) * 2) * FRAG_ELEMS + lane * 16;
    v16bf a0 = *(const v16bf*)(mkp + base);
    v16bf a1 = *(const v16bf*)(mkp + base + FRAG_ELEMS);
    v8f acc = {};
    acc = WMMA_BF16(a0, bq0, acc);
    acc = WMMA_BF16(a1, bq1, acc);

    const int t0 = si * 16;
    float m_new = m_run;
    float sc[8];
#pragma unroll
    for (int r = 0; r < 8; ++r) {
      sc[r] = 0.25f * acc[r] - 0.125f * msqb[t0 + r + 8 * half];
      m_new = fmaxf(m_new, sc[r]);
    }
    float sa = s_run * __expf(m_run - m_new);
#pragma unroll
    for (int r = 0; r < 8; ++r) sa += __expf(sc[r] - m_new);
    m_run = m_new; s_run = sa;
  }

  __shared__ float lm[256], ls[256];
  lm[threadIdx.x] = m_run; ls[threadIdx.x] = s_run;
  __syncthreads();
  if (threadIdx.x < QT) {
    int q = threadIdx.x;
    float M = -3.0e38f, S = 0.f;
#pragma unroll
    for (int i = 0; i < 16; ++i) {              // 8 waves x 2 lane-halves
      int idx = (i >> 1) * 32 + (i & 1) * 16 + q;
      float mm = lm[idx], ss = ls[idx];
      float Mn = fmaxf(M, mm);
      S = S * __expf(M - Mn) + ss * __expf(mm - Mn);
      M = Mn;
    }
    size_t o = ((size_t)b * NQT + qt) * QT + q;
    stat_m[o] = M;
    stat_s[o] = fmaxf(S, 1e-30f);
  }
}

// ---------------------------------------------------------------------------
// Kernel 4: output pass. grid = (NQT, B), 8 waves.
// Waves 0-3 rebuild S each 64-t chunk, write exp(S-m) bf16 into the
// transposed LDS tile P[q][row]; all waves consume P as B fragments.
// mv fragments stream via TDM into per-wave LDS double buffers (8 KB each),
// DMA of chunk i+1 overlapped with WMMA on chunk i; s_wait_tensorcnt syncs.
// ---------------------------------------------------------------------------
__global__ __launch_bounds__(256) void pass_out(const __bf16* __restrict__ mkp,
                                                const __bf16* __restrict__ mvp,
                                                const float* __restrict__ qk,
                                                const float* __restrict__ msq,
                                                const float* __restrict__ stat_m,
                                                const float* __restrict__ stat_s,
                                                float* __restrict__ out) {
  const int qt = blockIdx.x, b = blockIdx.y;
  const int lane = threadIdx.x & 31, w = threadIdx.x >> 5;
  const int qcol = lane & 15, half = lane >> 4;
  const int q0 = qt * QT;
  const int qg = min(q0 + qcol, HW - 1);

  // P[q][row] bf16, stride 80 keeps every v16bf access 32B-aligned (v8bf 16B).
  __shared__ __attribute__((aligned(32))) __bf16 P[16][80];
#if USE_TDM
  // Per-wave double buffer: 4 c-tiles x 2 fragments x 1024 B = 8 KB each.
  __shared__ __attribute__((aligned(64))) unsigned char MB[8][2][8192];
#endif

  const float* qkb  = qk  + (size_t)b * CK * HW;
  const float* msqb = msq + (size_t)b * THW;

  const size_t so = ((size_t)b * NQT + qt) * QT;
  const float mstat = stat_m[so + qcol];
  const float inv_s = 1.0f / stat_s[so + qcol];

  v16bf bq0, bq1;
#pragma unroll
  for (int e = 0; e < 16; ++e) {
    int c = 16 * half + e;
    bq0[e] = (__bf16)qkb[(size_t)c * HW + qg];
    bq1[e] = (__bf16)qkb[(size_t)(c + 32) * HW + qg];
  }

  v8f acc0 = {}, acc1 = {}, acc2 = {}, acc3 = {};

#if USE_TDM
  // Wave's mv pack rows start at c-tile (b*32 + w*4); row stride NTK_PAD frags.
  const unsigned long long mv_base = (unsigned long long)(uintptr_t)mvp;
  const unsigned long long row_bytes = (unsigned long long)NTK_PAD * 1024ull;
  const unsigned long long wave_base =
      mv_base + (unsigned long long)(b * 32 + w * 4) * row_bytes;
  // Prologue: DMA chunk 0 into buffer 0.
  tdm_load_2d((unsigned)(uintptr_t)&MB[w][0][0], wave_base + 0ull * 2048ull,
              /*tile_d0=*/256, /*tile_d1=*/4, /*row_stride=*/NTK_PAD * 128ull);
#endif

  for (int ci = 0; ci < NKT; ++ci) {
#if USE_TDM
    if (ci + 1 < NKT) {  // DMA next chunk while this one computes
      tdm_load_2d((unsigned)(uintptr_t)&MB[w][(ci + 1) & 1][0],
                  wave_base + (unsigned long long)(ci + 1) * 2048ull,
                  256, 4, NTK_PAD * 128ull);
    }
#endif

    // ---- S producers: waves 0-3 each own 16 t-rows of the 64-row chunk ----
    if (w < 4) {
      const int si = ci * 4 + w;
      if (si < NST) {                           // NST*16 == THW: rows all valid
        const size_t base = ((size_t)(b * NST + si) * 2) * FRAG_ELEMS + lane * 16;
        v16bf a0 = *(const v16bf*)(mkp + base);
        v16bf a1 = *(const v16bf*)(mkp + base + FRAG_ELEMS);
        v8f s = {};
        s = WMMA_BF16(a0, bq0, s);
        s = WMMA_BF16(a1, bq1, s);
        const int t0 = si * 16;
        v8bf pk;
#pragma unroll
        for (int r = 0; r < 8; ++r) {
          float sc = 0.25f * s[r] - 0.125f * msqb[t0 + r + 8 * half];
          pk[r] = (__bf16)__expf(sc - mstat);
        }
        *(v8bf*)&P[qcol][w * 16 + 8 * half] = pk;   // rows contiguous
      } else {                                  // tail of last chunk: P = 0
        v8bf z = {};
        *(v8bf*)&P[qcol][w * 16 + 8 * half] = z;
      }
    }
    __syncthreads();

    // ---- P as B operand: one 32B LDS vector load per K=32 fragment ----
    v16bf pb0 = *(const v16bf*)&P[qcol][16 * half];
    v16bf pb1 = *(const v16bf*)&P[qcol][32 + 16 * half];
    __syncthreads();                            // safe to overwrite next iter

    // ---- O accumulation: wave owns CV rows [w*64, w*64+64) ----
#if USE_TDM
    if (ci + 1 < NKT) __builtin_amdgcn_s_wait_tensorcnt((short)1);
    else              __builtin_amdgcn_s_wait_tensorcnt((short)0);
    const __bf16* buf = (const __bf16*)&MB[w][ci & 1][0];
#pragma unroll
    for (int ct = 0; ct < 4; ++ct) {
      v16bf av0 = *(const v16bf*)(buf + ct * 1024 + lane * 16);
      v16bf av1 = *(const v16bf*)(buf + ct * 1024 + FRAG_ELEMS + lane * 16);
      v8f* accp = (ct == 0) ? &acc0 : (ct == 1) ? &acc1 : (ct == 2) ? &acc2 : &acc3;
      *accp = WMMA_BF16(av0, pb0, *accp);
      *accp = WMMA_BF16(av1, pb1, *accp);       // tail: pb1==0, pad frag inert
    }
#else
    const int tk = 2 * ci;
#pragma unroll
    for (int ct = 0; ct < 4; ++ct) {
      const size_t fb =
          ((size_t)(b * 32 + w * 4 + ct) * NTK_PAD + tk) * FRAG_ELEMS + lane * 16;
      if (ci + 1 < NKT) __builtin_prefetch(mvp + fb + 2 * FRAG_ELEMS, 0, 1);
      v16bf av0 = *(const v16bf*)(mvp + fb);
      v16bf av1 = *(const v16bf*)(mvp + fb + FRAG_ELEMS);
      v8f* accp = (ct == 0) ? &acc0 : (ct == 1) ? &acc1 : (ct == 2) ? &acc2 : &acc3;
      *accp = WMMA_BF16(av0, pb0, *accp);
      *accp = WMMA_BF16(av1, pb1, *accp);
    }
#endif
  }

  // ---- store: D layout lane -> col q = qcol, VGPR r -> row c = base+r+8h ----
  const int q = q0 + qcol;
  if (q < HW) {
    v8f accs[4] = {acc0, acc1, acc2, acc3};
#pragma unroll
    for (int ct = 0; ct < 4; ++ct) {
#pragma unroll
      for (int r = 0; r < 8; ++r) {
        int c = w * 64 + ct * 16 + r + 8 * half;
        out[((size_t)b * (2 * CV) + c) * HW + q] = accs[ct][r] * inv_s;
      }
    }
  }
}

// ---------------------------------------------------------------------------
// Kernel 5: out[:, CV:2CV] = qv
// ---------------------------------------------------------------------------
__global__ __launch_bounds__(256) void copy_qv(const float* __restrict__ qv,
                                               float* __restrict__ out) {
  size_t i = (size_t)blockIdx.x * 256 + threadIdx.x;
  const size_t per_b = (size_t)CV * HW;
  if (i >= (size_t)BB * per_b) return;
  size_t b = i / per_b, rem = i - b * per_b;
  out[b * 2 * per_b + per_b + rem] = qv[i];
}

// ---------------------------------------------------------------------------
extern "C" void kernel_launch(void* const* d_in, const int* in_sizes, int n_in,
                              void* d_out, int out_size, void* d_ws, size_t ws_size,
                              hipStream_t stream) {
  const float* mk = (const float*)d_in[0];
  const float* qk = (const float*)d_in[1];
  const float* mv = (const float*)d_in[2];
  const float* qv = (const float*)d_in[3];
  float* out = (float*)d_out;

  // Workspace layout (all offsets 32B-aligned):
  //   msq     : BB*THW f32                       (207,360 B)
  //   stat_m  : BB*NQT*16 f32                    ( 26,112 B)
  //   stat_s  : BB*NQT*16 f32                    ( 26,112 B)
  //   mk_pack : BB*NST*2      frags * 1024 B     (  6.64 MB)
  //   mv_pack : BB*32*NTK_PAD frags * 1024 B     ( 53.2  MB)
  char* ws = (char*)d_ws;
  float*  msq    = (float*)ws;
  float*  stat_m = msq + BB * THW;
  float*  stat_s = stat_m + BB * NQT * QT;
  __bf16* mkp    = (__bf16*)(ws + 259584);
  __bf16* mvp    = (__bf16*)(ws + 259584 + (size_t)BB * NST * 2 * 1024);

  msq_kernel<<<(BB * THW + 255) / 256, 256, 0, stream>>>(mk, msq);
  pack_mk_kernel<<<(BB * NST * 2 * 32) / 256, 256, 0, stream>>>(mk, mkp);
  pack_mv_kernel<<<(BB * 32 * NTK_PAD * 32) / 256, 256, 0, stream>>>(mv, mvp);

  dim3 grid(NQT, BB);
  pass_stats<<<grid, 256, 0, stream>>>(mkp, qk, msq, stat_m, stat_s);
  pass_out<<<grid, 256, 0, stream>>>(mkp, mvp, qk, msq, stat_m, stat_s, out);
  copy_qv<<<((BB * (size_t)CV * HW) + 255) / 256, 256, 0, stream>>>(qv, out);
}
